// QuantizedLinear_68461778698493
// MI455X (gfx1250) — compile-verified, matrix operand order checked by reading, not run on hardware
//
#include <hip/hip_runtime.h>

// Quantized linear: out[b,s,o] = sum_i x[b,s,i] * (q[o,i]-zp[o])*scale[o] + bias[o]
// Decomposed as: y = scale[o]*(x @ q^T) - scale[o]*zp[o]*rowsum(x)[m] + bias[o]
// Heavy GEMM runs on v_wmma_f32_16x16x32_f16. int8 -> f16 uses the exact
// byte-perm magic trick: f16(0x6400 | (b^0x80)) - 1152 == b.
// K-step 64 (two WMMA k-chunks per LDS stage) halves barrier count vs k-step 32.

#define KD 4096
#define ND 11008
#define MD 8192
#define KSTEPS (KD / 64)          // 64 staged k-iterations, 2 WMMA chunks each
#define LDS_ROW 144               // 64 f16 (128B) + 16B pad, 16B aligned
#define LDS_TILE (128 * LDS_ROW)  // 18432 B per (A or B) tile
#define LDS_BUF (2 * LDS_TILE)    // A + B for one k-step

typedef __attribute__((ext_vector_type(16))) _Float16 v16h;
typedef __attribute__((ext_vector_type(2)))  _Float16 v2h;
typedef __attribute__((ext_vector_type(8)))  float    v8f;

union FragU  { v16h v; uint4 u[2]; };

// two signed bytes (selected by `sel` from xw = w ^ 0x80808080) -> packed f16 pair
__device__ __forceinline__ unsigned int i8pair2h(unsigned int xw, unsigned int sel) {
  unsigned int m = __builtin_amdgcn_perm(0x64646464u, xw, sel); // halfwords 0x64uu
  v2h h = __builtin_bit_cast(v2h, m) + (v2h){(_Float16)-1152.f, (_Float16)-1152.f};
  return __builtin_bit_cast(unsigned int, h);
}

// 16 int8 (uint4) -> 16 f16 packed in two uint4
__device__ __forceinline__ void cvt16i8(const uint4& br, uint4& p0, uint4& p1) {
  unsigned int w[4] = {br.x, br.y, br.z, br.w};
  unsigned int d[8];
#pragma unroll
  for (int q = 0; q < 4; ++q) {
    unsigned int xw = w[q] ^ 0x80808080u;
    d[q * 2 + 0] = i8pair2h(xw, 0x04010400u);  // bytes 0,1
    d[q * 2 + 1] = i8pair2h(xw, 0x04030402u);  // bytes 2,3
  }
  p0.x = d[0]; p0.y = d[1]; p0.z = d[2]; p0.w = d[3];
  p1.x = d[4]; p1.y = d[5]; p1.z = d[6]; p1.w = d[7];
}

// 16 f32 (four float4) -> 16 f16 packed in two uint4 (v_cvt_pk_f16_f32)
__device__ __forceinline__ void cvt16f32(const float4* ar, uint4& p0, uint4& p1) {
  union { _Float16 h[16]; uint4 u[2]; } s;
#pragma unroll
  for (int j = 0; j < 4; ++j) {
    s.h[j * 4 + 0] = (_Float16)ar[j].x;  s.h[j * 4 + 1] = (_Float16)ar[j].y;
    s.h[j * 4 + 2] = (_Float16)ar[j].z;  s.h[j * 4 + 3] = (_Float16)ar[j].w;
  }
  p0 = s.u[0]; p1 = s.u[1];
}

// ---- pre-pass: rowsum[m] = sum_k x[m,k], one wave32 per row ----------------
__global__ void __launch_bounds__(256)
qlin_rowsum(const float* __restrict__ x, float* __restrict__ rs) {
  const int row  = blockIdx.x * 8 + (threadIdx.x >> 5);
  const int lane = threadIdx.x & 31;
  const float4* p = (const float4*)(x + (size_t)row * KD);
  float s = 0.f;
  for (int i = lane; i < KD / 4; i += 32) {
    float4 v = p[i];
    s += (v.x + v.y) + (v.z + v.w);
  }
#pragma unroll
  for (int off = 16; off > 0; off >>= 1) s += __shfl_down(s, off, 32);
  if (lane == 0) rs[row] = s;
}

// ---- main GEMM: 128x128 tile / workgroup, 8 waves, 4x2 WMMA tiles / wave ---
__global__ void __launch_bounds__(256)
qlin_gemm(const float* __restrict__ x, const signed char* __restrict__ wq,
          const float* __restrict__ scale, const float* __restrict__ zp,
          const float* __restrict__ bias, const float* __restrict__ rowsum,
          float* __restrict__ out) {
  __shared__ __align__(16) char lds[2 * LDS_BUF];   // double-buffered A|B tiles (72 KB)

  const int tid   = threadIdx.x;
  const int lane  = tid & 31;
  const int wave  = tid >> 5;        // 0..7
  const int waveM = wave >> 2;       // 0..1  (64 rows each)
  const int waveN = wave & 3;        // 0..3  (32 cols each)
  const int half  = lane >> 4;       // K-half select per ISA A/B layout
  const int r     = lane & 15;       // row-within-16 / N-within-16

  const int bM = blockIdx.y * 128;
  const int bN = blockIdx.x * 128;

  // global staging: 128 rows x 2 chunks of 32 k-elements per thread
  const int lrow   = tid >> 1;       // 0..127
  const int c0     = (tid & 1) * 32; // k-chunk base within the 64-wide stage
  const float*       aG = x  + (size_t)(bM + lrow) * KD + c0;
  const signed char* bG = wq + (size_t)(bN + lrow) * KD + c0;

  char* aS = lds + lrow * LDS_ROW + c0 * 2;                    // + buf*LDS_BUF
  char* bS = lds + LDS_TILE + lrow * LDS_ROW + c0 * 2;

  // fragment bases (ISA 16-bit 16x32 layout: lanes0-15 K{0..7,16..23}, lanes16-31 K{8..15,24..31})
  const char* aF = lds + (waveM * 64 + r) * LDS_ROW + half * 16;
  const char* bF = lds + LDS_TILE + (waveN * 32 + r) * LDS_ROW + half * 16;

  v8f acc[4][2];
#pragma unroll
  for (int mi = 0; mi < 4; ++mi)
#pragma unroll
    for (int ni = 0; ni < 2; ++ni) acc[mi][ni] = (v8f){0.f,0.f,0.f,0.f,0.f,0.f,0.f,0.f};

  float4 ar[8];
  uint4  br[2];
  // prologue: stage k-tile 0 into buffer 0
#pragma unroll
  for (int j = 0; j < 8; ++j) ar[j] = *(const float4*)(aG + j * 4);
  br[0] = *(const uint4*)bG;
  br[1] = *(const uint4*)(bG + 16);
  {
    uint4 a0, a1, a2, a3, b0, b1, b2, b3;
    cvt16f32(ar, a0, a1);  cvt16f32(ar + 4, a2, a3);
    cvt16i8(br[0], b0, b1); cvt16i8(br[1], b2, b3);
    *(uint4*)aS = a0; *(uint4*)(aS + 16) = a1; *(uint4*)(aS + 32) = a2; *(uint4*)(aS + 48) = a3;
    *(uint4*)bS = b0; *(uint4*)(bS + 16) = b1; *(uint4*)(bS + 32) = b2; *(uint4*)(bS + 48) = b3;
  }
  __syncthreads();

  for (int kt = 0; kt < KSTEPS; ++kt) {
    const int buf = kt & 1;
    if (kt + 1 < KSTEPS) {           // prefetch next k-stage while WMMAs run
      const float* ap = aG + (size_t)(kt + 1) * 64;
#pragma unroll
      for (int j = 0; j < 8; ++j) ar[j] = *(const float4*)(ap + j * 4);
      br[0] = *(const uint4*)(bG + (size_t)(kt + 1) * 64);
      br[1] = *(const uint4*)(bG + (size_t)(kt + 1) * 64 + 16);
    }

    const char* aFb = aF + buf * LDS_BUF;
    const char* bFb = bF + buf * LDS_BUF;
#pragma unroll
    for (int kc = 0; kc < 2; ++kc) {   // two 32-wide WMMA chunks per stage
      FragU a[4], b[2];
#pragma unroll
      for (int ni = 0; ni < 2; ++ni) {
        const char* p = bFb + ni * (16 * LDS_ROW) + kc * 64;
        b[ni].u[0] = *(const uint4*)p;
        b[ni].u[1] = *(const uint4*)(p + 32);
      }
#pragma unroll
      for (int mi = 0; mi < 4; ++mi) {
        const char* p = aFb + mi * (16 * LDS_ROW) + kc * 64;
        a[mi].u[0] = *(const uint4*)p;
        a[mi].u[1] = *(const uint4*)(p + 32);
      }
#pragma unroll
      for (int mi = 0; mi < 4; ++mi)
#pragma unroll
        for (int ni = 0; ni < 2; ++ni)
          acc[mi][ni] = __builtin_amdgcn_wmma_f32_16x16x32_f16(
              false, a[mi].v, false, b[ni].v, (short)0, acc[mi][ni], false, false);
    }

    if (kt + 1 < KSTEPS) {           // convert + stage into the other buffer
      char* aSb = aS + (buf ^ 1) * LDS_BUF;
      char* bSb = bS + (buf ^ 1) * LDS_BUF;
      uint4 a0, a1, a2, a3, b0, b1, b2, b3;
      cvt16f32(ar, a0, a1);  cvt16f32(ar + 4, a2, a3);
      cvt16i8(br[0], b0, b1); cvt16i8(br[1], b2, b3);
      *(uint4*)aSb = a0; *(uint4*)(aSb + 16) = a1; *(uint4*)(aSb + 32) = a2; *(uint4*)(aSb + 48) = a3;
      *(uint4*)bSb = b0; *(uint4*)(bSb + 16) = b1; *(uint4*)(bSb + 32) = b2; *(uint4*)(bSb + 48) = b3;
    }
    __syncthreads();
  }

  // epilogue: y = sc*acc - sc*zp*rowsum[m] + bias ; C layout: M=r'+8*half, N=lane&15
  int   nn[2];
  float sc[2], szp[2], bs[2];
#pragma unroll
  for (int ni = 0; ni < 2; ++ni) {
    nn[ni]  = bN + waveN * 32 + ni * 16 + r;
    sc[ni]  = scale[nn[ni]];
    szp[ni] = sc[ni] * zp[nn[ni]];
    bs[ni]  = bias[nn[ni]];
  }
#pragma unroll
  for (int mi = 0; mi < 4; ++mi) {
    const int mb = bM + waveM * 64 + mi * 16 + half * 8;     // 8-aligned
    const float4 r0 = *(const float4*)(rowsum + mb);
    const float4 r1 = *(const float4*)(rowsum + mb + 4);
    const float rsv[8] = {r0.x, r0.y, r0.z, r0.w, r1.x, r1.y, r1.z, r1.w};
#pragma unroll
    for (int ni = 0; ni < 2; ++ni)
#pragma unroll
      for (int rr = 0; rr < 8; ++rr) {
        const int m = mb + rr;
        const float res = acc[mi][ni][rr] * sc[ni] - szp[ni] * rsv[rr] + bs[ni];
        __builtin_nontemporal_store(res, &out[(size_t)m * ND + nn[ni]]);
      }
  }
}

extern "C" void kernel_launch(void* const* d_in, const int* in_sizes, int n_in,
                              void* d_out, int out_size, void* d_ws, size_t ws_size,
                              hipStream_t stream) {
  const float*       x     = (const float*)d_in[0];
  const signed char* wq    = (const signed char*)d_in[1];  // int8 weights
  const float*       scale = (const float*)d_in[2];
  const float*       zp    = (const float*)d_in[3];
  const float*       bias  = (const float*)d_in[4];
  float*             out   = (float*)d_out;
  float*             rs    = (float*)d_ws;                 // MD floats = 32 KB

  qlin_rowsum<<<MD / 8, 256, 0, stream>>>(x, rs);

  dim3 grid(ND / 128, MD / 128);   // 86 x 64 workgroups
  qlin_gemm<<<grid, dim3(256), 0, stream>>>(x, wq, scale, zp, bias, rs, out);

  (void)in_sizes; (void)n_in; (void)out_size; (void)ws_size;
}
